// HQQLinearGroupedProj_39943195853111
// MI455X (gfx1250) — matrix operand order repaired and checked by reading, not compiled, not run on hardware
//
#include <hip/hip_runtime.h>
#include <stdint.h>

// ---------------- problem constants (match reference) ----------------
#define IN_F   4096
#define OUT_F  4096
#define MTOT   8192          // BATCH * SEQ = 4 * 2048
#define QGROUP 64
#define NGRP   (IN_F / QGROUP)   // 64
#define PNUM   128
#define PSZ    32

// ---------------- vector types for WMMA ----------------
typedef __bf16 bf16_t;
typedef bf16_t bf16x4  __attribute__((ext_vector_type(4)));
typedef bf16_t bf16x8  __attribute__((ext_vector_type(8)));
typedef bf16_t bf16x16 __attribute__((ext_vector_type(16)));
typedef float  f32x8   __attribute__((ext_vector_type(8)));

union ABFrag {
    bf16x16 v16;
    bf16x8  v8[2];
};

// =====================================================================
// Kernel 0: convert x (f32) -> bf16 workspace, 4 elements / thread
// =====================================================================
__global__ __launch_bounds__(256) void cvt_x_bf16(const float* __restrict__ x,
                                                  bf16_t* __restrict__ xb) {
    const size_t i = (size_t)blockIdx.x * blockDim.x + threadIdx.x; // float4 idx
    const float4 v = ((const float4*)x)[i];
    bf16x4 o;
    o[0] = (bf16_t)v.x; o[1] = (bf16_t)v.y;
    o[2] = (bf16_t)v.z; o[3] = (bf16_t)v.w;
    *(bf16x4*)(xb + 4 * i) = o;
}

// =====================================================================
// Kernel 1: W' = proj[p] @ ((W_q - zero) * scale), emitted as bf16.
// =====================================================================
__global__ __launch_bounds__(256) void dequant_proj(
        const int*   __restrict__ Wq,
        const float* __restrict__ scale,
        const float* __restrict__ zero,
        const float* __restrict__ proj,
        bf16_t*      __restrict__ Wp) {
    const int p = blockIdx.y;
    const int c = blockIdx.x * 256 + threadIdx.x;
    const int g = c >> 6;                      // quant group of this column

    __shared__ float pr[PSZ * PSZ];
    for (int i = threadIdx.x; i < PSZ * PSZ; i += 256)
        pr[i] = proj[p * (PSZ * PSZ) + i];
    __syncthreads();

    float w[PSZ];
#pragma unroll
    for (int j = 0; j < PSZ; ++j) {
        const int row = p * PSZ + j;
        const float s = scale[row * NGRP + g];
        const float z = zero [row * NGRP + g];
        w[j] = ((float)Wq[(size_t)row * IN_F + c] - z) * s;
    }

#pragma unroll 4
    for (int i = 0; i < PSZ; ++i) {
        float acc = 0.f;
#pragma unroll
        for (int j = 0; j < PSZ; ++j)
            acc = fmaf(pr[i * PSZ + j], w[j], acc);
        Wp[(size_t)(p * PSZ + i) * IN_F + c] = (bf16_t)acc;
    }
}

// =====================================================================
// Kernel 2: out = Xb(8192x4096) * Wp^T(4096x4096) + bias,  bf16 WMMA.
// Block tile 128x128, K-step 32. 256 threads = 8 wave32s (4x2); each
// wave owns a 32x64 sub-tile = 2x4 WMMA accumulators.
//
// LDS: single flat array; four 16 KiB buffers at fixed offsets
//   A0 @ 0x0000, A1 @ 0x4000, B0 @ 0x8000, B1 @ 0xC000
// Double buffering = XOR-toggling read/write offsets (parity-free body,
// #pragma unroll 1) so accumulators stay pinned across the back edge.
// Tiles staged with CDNA5 GLOBAL_LOAD_ASYNC_TO_LDS_B128 (ASYNCcnt),
// fenced with s_wait_asynccnt before the workgroup barrier.
// =====================================================================
#define BM  128
#define BN  128
#define BK  32
#define LDA 40                 // padded row stride (elements) -> 80 B
#define BUFE 8192              // elements per 16 KiB buffer
#define NK  (IN_F / BK)        // 128

__global__ __launch_bounds__(256) void gemm_bf16(
        const bf16_t* __restrict__ Xb,
        const bf16_t* __restrict__ Wp,
        const float*  __restrict__ bias,
        float*        __restrict__ out) {
    __shared__ __align__(16) unsigned short lds[4 * BUFE];   // 64 KiB

    const int tid  = threadIdx.x;
    const int lane = tid & 31;
    const int wave = tid >> 5;
    const int wm   = wave >> 1;     // 0..3 -> M offset wm*32
    const int wn   = wave & 1;      // 0..1 -> N offset wn*64
    const int h    = lane >> 4;     // lane half (K-split per ISA layout)
    const int r    = lane & 15;     // row/col within 16

    const int mBase = blockIdx.y * BM;
    const int nBase = blockIdx.x * BN;

    // async staging: thread t owns one 32-byte slot of the A tile and
    // one of the B tile (two b128 asyncs each, offset:0 / offset:16)
    const int trow = tid >> 1;            // 0..127
    const int tcol = (tid & 1) * 16;      // 0 or 16 (elements)
    const bf16_t* gA = Xb + (size_t)(mBase + trow) * IN_F + tcol;
    const bf16_t* gB = Wp + (size_t)(nBase + trow) * IN_F + tcol;

    // LDS byte addresses (generic->LDS: low 32 bits = LDS offset)
    const uint32_t ldsBase = (uint32_t)(uintptr_t)&lds[0];
    const uint32_t thrSlot = (uint32_t)(trow * LDA + tcol) * 2u;  // bytes
    // prologue writes buffer 0; steady-state writes start at buffer 1
    uint32_t wrA = ldsBase + 0x4000u + thrSlot;
    uint32_t wrB = ldsBase + 0xC000u + thrSlot;
    // read element-offsets into flat lds[], toggled by ^BUFE
    int rdA = 0;
    int rdB = 2 * BUFE;

    f32x8 acc[2][4];
    const f32x8 zero8 = {};
#pragma unroll
    for (int m = 0; m < 2; ++m)
#pragma unroll
        for (int n = 0; n < 4; ++n) acc[m][n] = zero8;

    // ---- prologue: async-stage K-tile 0 into buffer 0 ----
    {
        const uint64_t ga = (uint64_t)(uintptr_t)gA;
        const uint64_t gb = (uint64_t)(uintptr_t)gB;
        const uint32_t a0 = ldsBase + thrSlot;            // A0
        const uint32_t b0 = ldsBase + 0x8000u + thrSlot;  // B0
        asm volatile(
            "global_load_async_to_lds_b128 %0, %2, off\n\t"
            "global_load_async_to_lds_b128 %0, %2, off offset:16\n\t"
            "global_load_async_to_lds_b128 %1, %3, off\n\t"
            "global_load_async_to_lds_b128 %1, %3, off offset:16"
            :
            : "v"(a0), "v"(b0), "v"(ga), "v"(gb)
            : "memory");
        asm volatile("s_wait_asynccnt 0" ::: "memory");
    }
    __syncthreads();

    const unsigned short* ldsFlat = &lds[0];
    const int aRow0 = (wm * 32 + r) * LDA;   // + m*16*LDA per m-frag
    const int bRow0 = (wn * 64 + r) * LDA;   // + n*16*LDA per n-frag

#pragma unroll 1
    for (int kt = 0; kt < NK; ++kt) {
        if (kt + 1 < NK) {   // async-stage next K-tile into the other buffer
            const uint64_t ga = (uint64_t)(uintptr_t)(gA + (size_t)(kt + 1) * BK);
            const uint64_t gb = (uint64_t)(uintptr_t)(gB + (size_t)(kt + 1) * BK);
            asm volatile(
                "global_load_async_to_lds_b128 %0, %2, off\n\t"
                "global_load_async_to_lds_b128 %0, %2, off offset:16\n\t"
                "global_load_async_to_lds_b128 %1, %3, off\n\t"
                "global_load_async_to_lds_b128 %1, %3, off offset:16"
                :
                : "v"(wrA), "v"(wrB), "v"(ga), "v"(gb)
                : "memory");
        }

        // fragment loads per CDNA5 16-bit WMMA VGPR layouts
        ABFrag a[2], b[4];
#pragma unroll
        for (int m = 0; m < 2; ++m) {
            const unsigned short* rowp = ldsFlat + rdA + aRow0 + m * (16 * LDA);
            // A: lane-half h holds K = h*8..h*8+7 and 16+h*8..16+h*8+7
            a[m].v8[0] = *(const bf16x8*)(rowp + h * 8);
            a[m].v8[1] = *(const bf16x8*)(rowp + 16 + h * 8);
        }
#pragma unroll
        for (int n = 0; n < 4; ++n) {
            const unsigned short* rowp = ldsFlat + rdB + bRow0 + n * (16 * LDA);
            // B: lane-half h holds K = h*8..h*8+7 and 16+h*8..16+h*8+7
            b[n].v8[0] = *(const bf16x8*)(rowp + h * 8);
            b[n].v8[1] = *(const bf16x8*)(rowp + 16 + h * 8);
        }

#pragma unroll
        for (int m = 0; m < 2; ++m)
#pragma unroll
            for (int n = 0; n < 4; ++n)
                acc[m][n] = __builtin_amdgcn_wmma_f32_16x16x32_bf16(
                    /*neg_a=*/false, a[m].v16,
                    /*neg_b=*/false, b[n].v16,
                    /*c_mod=*/(short)0, acc[m][n],
                    /*reuse_a=*/false, /*reuse_b=*/false);

        if (kt + 1 < NK)   // our async fills must land before anyone reads them
            asm volatile("s_wait_asynccnt 0" ::: "memory");
        __syncthreads();

        // toggle double buffers (16 KiB apart)
        rdA ^= BUFE;
        rdB ^= BUFE;
        wrA ^= 0x4000u;
        wrB ^= 0x4000u;
    }

    // epilogue: D layout -> M = v + 8*h, N = r ; fuse bias add
#pragma unroll
    for (int n = 0; n < 4; ++n) {
        const int col = nBase + wn * 64 + n * 16 + r;
        const float bv = bias[col];
#pragma unroll
        for (int m = 0; m < 2; ++m) {
            const int row0 = mBase + wm * 32 + m * 16 + 8 * h;
#pragma unroll
            for (int v = 0; v < 8; ++v)
                out[(size_t)(row0 + v) * OUT_F + col] = acc[m][n][v] + bv;
        }
    }
}

// =====================================================================
// launch
// =====================================================================
extern "C" void kernel_launch(void* const* d_in, const int* in_sizes, int n_in,
                              void* d_out, int out_size, void* d_ws, size_t ws_size,
                              hipStream_t stream) {
    const float* x     = (const float*)d_in[0];
    const int*   Wq    = (const int*)  d_in[1];
    const float* scale = (const float*)d_in[2];
    const float* zero  = (const float*)d_in[3];
    const float* proj  = (const float*)d_in[4];
    const float* bias  = (const float*)d_in[5];
    float*       out   = (float*)d_out;

    // workspace: [ Xb bf16 : 8192*4096*2 = 64 MiB ][ Wp bf16 : 32 MiB ]
    bf16_t* Xb = (bf16_t*)d_ws;
    bf16_t* Wp = (bf16_t*)((char*)d_ws + (size_t)MTOT * IN_F * sizeof(bf16_t));

    // 0) x f32 -> bf16 (33.55M elems, 4/thread)
    cvt_x_bf16<<<dim3((MTOT * (size_t)IN_F) / 4 / 256), dim3(256), 0, stream>>>(x, Xb);

    // 1) dequant + grouped projection -> Wp bf16
    dequant_proj<<<dim3(IN_F / 256, PNUM), dim3(256), 0, stream>>>(
        Wq, scale, zero, proj, Wp);

    // 2) main GEMM + bias on bf16 WMMA
    gemm_bf16<<<dim3(OUT_F / BN, MTOT / BM), dim3(256), 0, stream>>>(
        Xb, Wp, bias, out);
}
